// Large_to_Mini_46961172414969
// MI455X (gfx1250) — matrix operand order repaired and checked by reading, not compiled, not run on hardware
//
#include <hip/hip_runtime.h>

typedef __attribute__((ext_vector_type(16))) _Float16 v16h;
typedef __attribute__((ext_vector_type(8)))  _Float16 v8h;
typedef __attribute__((ext_vector_type(8)))  float    v8f;
typedef __attribute__((ext_vector_type(4)))  float    v4f;

#define NPTS     1048576
#define NGROUPS  (NPTS / 16)
#define BLOCK    128
#define WAVES    (BLOCK / 32)

__global__ __launch_bounds__(BLOCK) void fused_gather_mlp(
    const float* __restrict__ features,   // [16384, 64]
    const int*   __restrict__ labels,     // [NPTS]
    const float* __restrict__ centers,    // [16384, 3]
    const float* __restrict__ points,     // [NPTS, 3]
    const float* __restrict__ W1,         // [67, 128]
    const float* __restrict__ b1,         // [128]
    const float* __restrict__ W2,         // [128, 64]
    const float* __restrict__ b2,         // [64]
    float* __restrict__ out)              // [NPTS, 64]
{
    // Weights transposed to [N][K] in f16 so each lane's B-fragment is a
    // contiguous 32B read; K of layer 1 padded 67 -> 96.
    __shared__ __attribute__((aligned(32))) _Float16 sW1[128 * 96];
    __shared__ __attribute__((aligned(32))) _Float16 sW2[64 * 128];
    __shared__ float sB1[128];
    __shared__ float sB2[64];
    // Per-wave h staging tile (C/D -> A-layout transpose goes through LDS).
    __shared__ __attribute__((aligned(32))) _Float16 sH[WAVES][16 * 128];

    const int tid = threadIdx.x;

    // ---- One-time per-block weight conversion (f32 -> f16, transpose) ----
    for (int i = tid; i < 128 * 96; i += BLOCK) {
        int n = i / 96, k = i % 96;
        sW1[i] = (k < 67) ? (_Float16)W1[k * 128 + n] : (_Float16)0.0f;
    }
    for (int i = tid; i < 64 * 128; i += BLOCK) {
        int n = i >> 7, k = i & 127;
        sW2[i] = (_Float16)W2[k * 64 + n];
    }
    if (tid < 128) sB1[tid] = b1[tid];
    if (tid < 64)  sB2[tid] = b2[tid];
    __syncthreads();

    const int wave = tid >> 5;
    const int lane = tid & 31;
    const int m    = lane & 15;     // A row / B,C,D column within 16x16 tile
    const int hi   = lane >> 4;     // half-wave select
    const int kbA  = hi * 16;       // A-fragment K sub-offset (per 32-K step)
    const int kbB  = hi * 16;       // B-fragment K sub-offset
    _Float16* __restrict__ sHw = sH[wave];

    for (int g = blockIdx.x * WAVES + wave; g < NGROUPS; g += gridDim.x * WAVES) {
        const int row0 = g * 16;
        const int row  = row0 + m;
        const int lab  = labels[row];
        const float* feat = features + (size_t)lab * 64;

        const float d0 = centers[lab * 3 + 0] - points[(size_t)row * 3 + 0];
        const float d1 = centers[lab * 3 + 1] - points[(size_t)row * 3 + 1];
        const float d2 = centers[lab * 3 + 2] - points[(size_t)row * 3 + 2];

        // ---- A-fragments for GEMM1: x row = [feat[0:64] | d | 0-pad] ----
        // 16-bit A 16x32 layout: lane m holds K {kbA..kbA+15} (contiguous, matches B)
        const v4f* f4 = (const v4f*)feat;
        v4f t0 = f4[kbA / 4 + 0],  t1 = f4[kbA / 4 + 1];    // K-step 0, elems 0-7
        v4f t2 = f4[kbA / 4 + 2],  t3 = f4[kbA / 4 + 3];    // K-step 0, elems 8-15
        v4f t4 = f4[kbA / 4 + 8],  t5 = f4[kbA / 4 + 9];    // K-step 1, elems 0-7
        v4f t6 = f4[kbA / 4 + 10], t7 = f4[kbA / 4 + 11];   // K-step 1, elems 8-15

        v16h a0, a1, a2 = {};
#pragma unroll
        for (int j = 0; j < 4; j++) {
            a0[j]      = (_Float16)t0[j];  a0[j + 4]  = (_Float16)t1[j];
            a0[j + 8]  = (_Float16)t2[j];  a0[j + 12] = (_Float16)t3[j];
            a1[j]      = (_Float16)t4[j];  a1[j + 4]  = (_Float16)t5[j];
            a1[j + 8]  = (_Float16)t6[j];  a1[j + 12] = (_Float16)t7[j];
        }
        // K-step 2 holds K=64..66 (coord diffs) in lanes 0-15 only, rest zero.
        a2[0] = hi ? (_Float16)0.0f : (_Float16)d0;
        a2[1] = hi ? (_Float16)0.0f : (_Float16)d1;
        a2[2] = hi ? (_Float16)0.0f : (_Float16)d2;

        // ---- GEMM1: h[16x128] = relu(x @ W1 + b1), h -> LDS as f16 ----
#pragma unroll
        for (int nt = 0; nt < 8; nt++) {
            const _Float16* wp = sW1 + (size_t)(nt * 16 + m) * 96 + kbB;
            v16h bv0 = *(const v16h*)(wp);
            v16h bv1 = *(const v16h*)(wp + 32);
            v16h bv2 = *(const v16h*)(wp + 64);
            v8f acc = {};
            acc = __builtin_amdgcn_wmma_f32_16x16x32_f16(false, a0, false, bv0, (short)0, acc, false, false);
            acc = __builtin_amdgcn_wmma_f32_16x16x32_f16(false, a1, false, bv1, (short)0, acc, false, false);
            acc = __builtin_amdgcn_wmma_f32_16x16x32_f16(false, a2, false, bv2, (short)0, acc, false, false);
            const float bias = sB1[nt * 16 + m];
#pragma unroll
            for (int i = 0; i < 8; i++) {
                float v = acc[i] + bias;
                v = v > 0.0f ? v : 0.0f;                  // relu
                sHw[(i + 8 * hi) * 128 + nt * 16 + m] = (_Float16)v;
            }
        }
        // Same-wave LDS ops are in order (DScnt); compiler inserts waits.

        // ---- A-fragments for GEMM2 from the h tile ----
        v16h ha[4];
        const _Float16* hp = sHw + (size_t)m * 128;
#pragma unroll
        for (int ks = 0; ks < 4; ks++) {
            v8h lo = *(const v8h*)(hp + 32 * ks + kbA);
            v8h hh = *(const v8h*)(hp + 32 * ks + kbA + 8);
#pragma unroll
            for (int j = 0; j < 8; j++) { ha[ks][j] = lo[j]; ha[ks][j + 8] = hh[j]; }
        }

        // ---- GEMM2: out[16x64] = relu(h @ W2 + b2) ----
#pragma unroll
        for (int nt = 0; nt < 4; nt++) {
            const _Float16* wp = sW2 + (size_t)(nt * 16 + m) * 128 + kbB;
            v8f acc = {};
#pragma unroll
            for (int ks = 0; ks < 4; ks++) {
                v16h bv = *(const v16h*)(wp + 32 * ks);
                acc = __builtin_amdgcn_wmma_f32_16x16x32_f16(false, ha[ks], false, bv, (short)0, acc, false, false);
            }
            const float bias = sB2[nt * 16 + m];
            float* op = out + (size_t)row0 * 64 + nt * 16 + m;
#pragma unroll
            for (int i = 0; i < 8; i++) {
                float v = acc[i] + bias;
                v = v > 0.0f ? v : 0.0f;                  // relu
                op[(size_t)(i + 8 * hi) * 64] = v;
            }
        }
    }
}

extern "C" void kernel_launch(void* const* d_in, const int* in_sizes, int n_in,
                              void* d_out, int out_size, void* d_ws, size_t ws_size,
                              hipStream_t stream) {
    const float* features = (const float*)d_in[0];
    const int*   labels   = (const int*)d_in[1];
    const float* centers  = (const float*)d_in[2];
    const float* points   = (const float*)d_in[3];
    const float* W1       = (const float*)d_in[4];
    const float* b1       = (const float*)d_in[5];
    const float* W2       = (const float*)d_in[6];
    const float* b2       = (const float*)d_in[7];
    float* out = (float*)d_out;

    // 4096 blocks x 4 waves = 16384 wave-slots; 65536 row-groups -> 4 each.
    dim3 grid(4096), block(BLOCK);
    fused_gather_mlp<<<grid, block, 0, stream>>>(features, labels, centers, points,
                                                 W1, b1, W2, b2, out);
}
